// JaxMoE_26431228740246
// MI455X (gfx1250) — matrix operand order repaired and checked by reading, not compiled, not run on hardware
//
#include <hip/hip_runtime.h>
#include <math.h>

// Problem constants (match reference)
#define T_TOK 2048
#define D_DIM 1024
#define F_DIM 4096
#define E_EXP 8

// GEMM tiling: 128x64 block tile, K-step 32, 8 wave32 waves (256 threads).
// Each wave computes a 32x32 tile = 2x2 fragments of 16x16 via
// v_wmma_f32_16x16x32_bf16.
#define BM 128
#define BN 64
#define BK 32
#define LSTR 40   // padded LDS row stride in bf16 elements (80 B, 16B-aligned)

typedef __attribute__((ext_vector_type(2)))  unsigned int u32x2;
typedef __attribute__((ext_vector_type(4)))  float        f32x4;
typedef __attribute__((ext_vector_type(4)))  unsigned int u32x4;
typedef __attribute__((ext_vector_type(8)))  float        v8f;
typedef __attribute__((ext_vector_type(16))) __bf16       v16bf;

union Frag {
    v16bf v;
    u32x4 q[2];
};

// Async LDS DMA path (gfx1250): pure-copy staging without VGPR round trip.
#if defined(__AMDGCN__) && __has_builtin(__builtin_amdgcn_global_load_async_to_lds_b128) && __has_builtin(__builtin_amdgcn_s_wait_asynccnt)
#define HAVE_ASYNC_LDS 1
typedef __attribute__((ext_vector_type(4))) int v4i;
typedef __attribute__((address_space(1))) v4i* gas_v4i_ptr;   // global (AS1) int4*
typedef __attribute__((address_space(3))) v4i* las_v4i_ptr;   // LDS (AS3) int4*
#else
#define HAVE_ASYNC_LDS 0
#endif

__device__ __forceinline__ unsigned short f2bf(float f) {
    unsigned int u = __builtin_bit_cast(unsigned int, f);
    u += 0x7fffu + ((u >> 16) & 1u);   // round-to-nearest-even
    return (unsigned short)(u >> 16);
}

// Pack two f32 into packed bf16x2 (one v_cvt_pk_bf16_f32 when available).
__device__ __forceinline__ unsigned int pack2bf(float a, float b) {
#if defined(__AMDGCN__) && __has_builtin(__builtin_amdgcn_cvt_pk_bf16_f32)
    auto r = __builtin_amdgcn_cvt_pk_bf16_f32(a, b);
    return __builtin_bit_cast(unsigned int, r);
#else
    return (unsigned int)f2bf(a) | ((unsigned int)f2bf(b) << 16);
#endif
}

// silu without IEEE-division expansion: one v_exp + one v_rcp.
__device__ __forceinline__ float fast_silu(float g) {
#if defined(__AMDGCN__) && __has_builtin(__builtin_amdgcn_rcpf)
    return g * __builtin_amdgcn_rcpf(1.0f + __expf(-g));
#else
    return g / (1.0f + __expf(-g));
#endif
}

__device__ __forceinline__ v16bf ld_frag(const unsigned short* p0,
                                         const unsigned short* p1) {
    Frag f;
    f.q[0] = *(const u32x4*)p0;   // ds_load_b128
    f.q[1] = *(const u32x4*)p1;   // ds_load_b128
    return f.v;
}

// ---------------------------------------------------------------------------
// Router: logits = x @ w_router, softmax, top-2, renormalize -> dense [T,E]
// Tiny (34 MFLOP): one thread per token.
// ---------------------------------------------------------------------------
__global__ __launch_bounds__(256) void moe_router(
    const float* __restrict__ x,        // [T, D]
    const float* __restrict__ wr,       // [D, E]
    float* __restrict__ wTE)            // [T, E] out
{
    int t = blockIdx.x * blockDim.x + threadIdx.x;
    if (t >= T_TOK) return;

    float logits[E_EXP];
#pragma unroll
    for (int e = 0; e < E_EXP; ++e) logits[e] = 0.0f;

    const float* xr = x + (size_t)t * D_DIM;
    for (int d = 0; d < D_DIM; ++d) {
        float xv = xr[d];
        const float* wrow = wr + (size_t)d * E_EXP;
#pragma unroll
        for (int e = 0; e < E_EXP; ++e) logits[e] += xv * wrow[e];
    }

    float m = logits[0];
#pragma unroll
    for (int e = 1; e < E_EXP; ++e) m = fmaxf(m, logits[e]);
    float p[E_EXP], s = 0.0f;
#pragma unroll
    for (int e = 0; e < E_EXP; ++e) { p[e] = __expf(logits[e] - m); s += p[e]; }
    float inv = 1.0f / s;
#pragma unroll
    for (int e = 0; e < E_EXP; ++e) p[e] *= inv;

    // top-2 (ties -> lowest index, matches lax.top_k)
    int i1 = 0;
#pragma unroll
    for (int e = 1; e < E_EXP; ++e) if (p[e] > p[i1]) i1 = e;
    int i2 = (i1 == 0) ? 1 : 0;
#pragma unroll
    for (int e = 0; e < E_EXP; ++e)
        if (e != i1 && p[e] > p[i2]) i2 = e;

    float z = 1.0f / (p[i1] + p[i2]);
    float* wout = wTE + (size_t)t * E_EXP;
#pragma unroll
    for (int e = 0; e < E_EXP; ++e)
        wout[e] = (e == i1) ? p[i1] * z : ((e == i2) ? p[i2] * z : 0.0f);
}

// ---------------------------------------------------------------------------
// Zero the f32 output accumulator.
// ---------------------------------------------------------------------------
__global__ __launch_bounds__(256) void zero_f32(float* __restrict__ p, int n) {
    int i = blockIdx.x * blockDim.x + threadIdx.x;
    if (i < n) p[i] = 0.0f;
}

// ---------------------------------------------------------------------------
// Pass 1 (per expert): h = router_w[t,e] * silu(x@Wg) * (x@Wu), stored bf16.
// M=T, N=F, K=D. f32 global -> bf16 LDS tiles -> WMMA bf16.
// ---------------------------------------------------------------------------
__global__ __launch_bounds__(256) void moe_gateup(
    const float* __restrict__ x,        // [T, D]
    const float* __restrict__ Wg,       // [D, F] (this expert)
    const float* __restrict__ Wu,       // [D, F] (this expert)
    const float* __restrict__ wTE,      // [T, E]
    unsigned short* __restrict__ h,     // [T, F] bf16 out
    int expert)
{
    __shared__ unsigned short ldsA[BM * LSTR];   // x tile   [m][k]
    __shared__ unsigned short ldsG[BN * LSTR];   // Wg tile  [n][k] (transposed)
    __shared__ unsigned short ldsU[BN * LSTR];   // Wu tile  [n][k]
    __shared__ float ldsW[BM];                   // router weights for tile rows

    const int tid = threadIdx.x;
    const int n0  = blockIdx.x * BN;
    const int m0  = blockIdx.y * BM;

    if (tid < BM) ldsW[tid] = wTE[(size_t)(m0 + tid) * E_EXP + expert];

    const int lane   = tid & 31;
    const int wave   = tid >> 5;
    const int wm     = (wave & 3) * 32;   // wave M offset in block tile
    const int wn     = (wave >> 2) * 32;  // wave N offset in block tile
    const int laneHi = lane >> 4;
    const int laneLo = lane & 15;

    v8f accG[2][2] = {};
    v8f accU[2][2] = {};

    for (int k0 = 0; k0 < D_DIM; k0 += BK) {
        __syncthreads();
        // Stage A: 128x32 f32 -> packed bf16, row-major [m][k], b64 stores
#pragma unroll
        for (int p = 0; p < 4; ++p) {
            int idx = tid + p * 256;
            int row = idx >> 3;
            int kc  = (idx & 7) * 4;
            f32x4 v = *(const f32x4*)(x + (size_t)(m0 + row) * D_DIM + k0 + kc);
            u32x2 pk;
            pk.x = pack2bf(v.x, v.y);
            pk.y = pack2bf(v.z, v.w);
            *(u32x2*)&ldsA[row * LSTR + kc] = pk;
        }
        // Stage B (gate & up): 32x64 f32, transposed into [n][k] bf16
#pragma unroll
        for (int p = 0; p < 2; ++p) {
            int idx = tid + p * 256;
            int k   = idx >> 4;
            int nc  = (idx & 15) * 4;
            f32x4 g = *(const f32x4*)(Wg + (size_t)(k0 + k) * F_DIM + n0 + nc);
            f32x4 u = *(const f32x4*)(Wu + (size_t)(k0 + k) * F_DIM + n0 + nc);
            unsigned int g01 = pack2bf(g.x, g.y), g23 = pack2bf(g.z, g.w);
            unsigned int u01 = pack2bf(u.x, u.y), u23 = pack2bf(u.z, u.w);
            ldsG[(nc + 0) * LSTR + k] = (unsigned short)g01;
            ldsG[(nc + 1) * LSTR + k] = (unsigned short)(g01 >> 16);
            ldsG[(nc + 2) * LSTR + k] = (unsigned short)g23;
            ldsG[(nc + 3) * LSTR + k] = (unsigned short)(g23 >> 16);
            ldsU[(nc + 0) * LSTR + k] = (unsigned short)u01;
            ldsU[(nc + 1) * LSTR + k] = (unsigned short)(u01 >> 16);
            ldsU[(nc + 2) * LSTR + k] = (unsigned short)u23;
            ldsU[(nc + 3) * LSTR + k] = (unsigned short)(u23 >> 16);
        }
        __syncthreads();

        v16bf af[2], gf[2], uf[2];
#pragma unroll
        for (int mi = 0; mi < 2; ++mi) {
            const unsigned short* r = &ldsA[(wm + mi * 16 + laneLo) * LSTR + laneHi * 8];
            af[mi] = ld_frag(r, r + 16);
        }
#pragma unroll
        for (int ni = 0; ni < 2; ++ni) {
            const unsigned short* rg = &ldsG[(wn + ni * 16 + laneLo) * LSTR + laneHi * 8];
            const unsigned short* ru = &ldsU[(wn + ni * 16 + laneLo) * LSTR + laneHi * 8];
            gf[ni] = ld_frag(rg, rg + 16);
            uf[ni] = ld_frag(ru, ru + 16);
        }
#pragma unroll
        for (int mi = 0; mi < 2; ++mi)
#pragma unroll
            for (int ni = 0; ni < 2; ++ni) {
                accG[mi][ni] = __builtin_amdgcn_wmma_f32_16x16x32_bf16(
                    false, af[mi], false, gf[ni], (short)0, accG[mi][ni], false, false);
                accU[mi][ni] = __builtin_amdgcn_wmma_f32_16x16x32_bf16(
                    false, af[mi], false, uf[ni], (short)0, accU[mi][ni], false, false);
            }
    }

    // Epilogue: h = w * silu(g) * u  (bf16 store, rows processed in pairs)
#pragma unroll
    for (int mi = 0; mi < 2; ++mi)
#pragma unroll
        for (int ni = 0; ni < 2; ++ni) {
            int gc = n0 + wn + ni * 16 + laneLo;
#pragma unroll
            for (int r = 0; r < 8; r += 2) {
                int lr0 = wm + mi * 16 + laneHi * 8 + r;
                int lr1 = lr0 + 1;
                float hv0 = fast_silu(accG[mi][ni][r])     * accU[mi][ni][r]     * ldsW[lr0];
                float hv1 = fast_silu(accG[mi][ni][r + 1]) * accU[mi][ni][r + 1] * ldsW[lr1];
                unsigned int pk = pack2bf(hv0, hv1);
                h[(size_t)(m0 + lr0) * F_DIM + gc] = (unsigned short)pk;
                h[(size_t)(m0 + lr1) * F_DIM + gc] = (unsigned short)(pk >> 16);
            }
        }
}

// ---------------------------------------------------------------------------
// Pass 2 (per expert): out += h(bf16) @ Wd.  M=T, N=D, K=F.
// A tile is a pure bf16 copy -> use gfx1250 async LDS DMA when available.
// ---------------------------------------------------------------------------
__global__ __launch_bounds__(256) void moe_down(
    const unsigned short* __restrict__ h,   // [T, F] bf16
    const float* __restrict__ Wd,           // [F, D] (this expert)
    float* __restrict__ out)                // [T, D] f32 accumulator
{
    __shared__ unsigned short ldsA[BM * LSTR];   // h tile  [m][k]
    __shared__ unsigned short ldsB[BN * LSTR];   // Wd tile [n][k]

    const int tid = threadIdx.x;
    const int n0  = blockIdx.x * BN;
    const int m0  = blockIdx.y * BM;

    const int lane   = tid & 31;
    const int wave   = tid >> 5;
    const int wm     = (wave & 3) * 32;
    const int wn     = (wave >> 2) * 32;
    const int laneHi = lane >> 4;
    const int laneLo = lane & 15;

    v8f acc[2][2] = {};

    for (int k0 = 0; k0 < F_DIM; k0 += BK) {
        __syncthreads();
        // Stage A: bf16 direct copy, 128x32 halves (8 KB)
#pragma unroll
        for (int p = 0; p < 2; ++p) {
            int idx = tid + p * 256;
            int row = idx >> 2;
            int kc  = (idx & 3) * 8;
            const unsigned short* gsrc = h + (size_t)(m0 + row) * F_DIM + k0 + kc;
            unsigned short* ldst = &ldsA[row * LSTR + kc];
#if HAVE_ASYNC_LDS
            __builtin_amdgcn_global_load_async_to_lds_b128(
                (gas_v4i_ptr)(void*)gsrc, (las_v4i_ptr)(void*)ldst, 0, 0);
#else
            *(u32x4*)ldst = *(const u32x4*)gsrc;
#endif
        }
        // Stage B: Wd 32x64 f32 -> transposed bf16 [n][k]
#pragma unroll
        for (int p = 0; p < 2; ++p) {
            int idx = tid + p * 256;
            int k   = idx >> 4;
            int nc  = (idx & 15) * 4;
            f32x4 b = *(const f32x4*)(Wd + (size_t)(k0 + k) * D_DIM + n0 + nc);
            unsigned int b01 = pack2bf(b.x, b.y), b23 = pack2bf(b.z, b.w);
            ldsB[(nc + 0) * LSTR + k] = (unsigned short)b01;
            ldsB[(nc + 1) * LSTR + k] = (unsigned short)(b01 >> 16);
            ldsB[(nc + 2) * LSTR + k] = (unsigned short)b23;
            ldsB[(nc + 3) * LSTR + k] = (unsigned short)(b23 >> 16);
        }
#if HAVE_ASYNC_LDS
        __builtin_amdgcn_s_wait_asynccnt(0);   // async LDS writes visible
#endif
        __syncthreads();

        v16bf af[2], bf[2];
#pragma unroll
        for (int mi = 0; mi < 2; ++mi) {
            const unsigned short* r = &ldsA[(wm + mi * 16 + laneLo) * LSTR + laneHi * 8];
            af[mi] = ld_frag(r, r + 16);
        }
#pragma unroll
        for (int ni = 0; ni < 2; ++ni) {
            const unsigned short* r = &ldsB[(wn + ni * 16 + laneLo) * LSTR + laneHi * 8];
            bf[ni] = ld_frag(r, r + 16);
        }
#pragma unroll
        for (int mi = 0; mi < 2; ++mi)
#pragma unroll
            for (int ni = 0; ni < 2; ++ni)
                acc[mi][ni] = __builtin_amdgcn_wmma_f32_16x16x32_bf16(
                    false, af[mi], false, bf[ni], (short)0, acc[mi][ni], false, false);
    }

    // Accumulate into out (experts run stream-sequentially; no races in-launch)
#pragma unroll
    for (int mi = 0; mi < 2; ++mi)
#pragma unroll
        for (int ni = 0; ni < 2; ++ni) {
            int gc = n0 + wn + ni * 16 + laneLo;
#pragma unroll
            for (int r = 0; r < 8; ++r) {
                int gr = m0 + wm + mi * 16 + laneHi * 8 + r;
                out[(size_t)gr * D_DIM + gc] += acc[mi][ni][r];
            }
        }
}

// ---------------------------------------------------------------------------
extern "C" void kernel_launch(void* const* d_in, const int* in_sizes, int n_in,
                              void* d_out, int out_size, void* d_ws, size_t ws_size,
                              hipStream_t stream) {
    (void)in_sizes; (void)n_in; (void)out_size; (void)ws_size;

    const float* x  = (const float*)d_in[0];   // [T, D]
    const float* wr = (const float*)d_in[1];   // [D, E]
    const float* Wg = (const float*)d_in[2];   // [E, D, F]
    const float* Wu = (const float*)d_in[3];   // [E, D, F]
    const float* Wd = (const float*)d_in[4];   // [E, F, D]
    float* out = (float*)d_out;                // [T, D]

    // Workspace: router weights f32 [T,E] (64 KB), then h bf16 [T,F] (16 MB)
    float* wTE = (float*)d_ws;
    unsigned short* hBuf =
        (unsigned short*)((char*)d_ws + (size_t)T_TOK * E_EXP * sizeof(float));

    moe_router<<<T_TOK / 256, 256, 0, stream>>>(x, wr, wTE);
    zero_f32<<<(T_TOK * D_DIM) / 256, 256, 0, stream>>>(out, T_TOK * D_DIM);

    dim3 grid1(F_DIM / BN, T_TOK / BM);   // 64 x 16
    dim3 grid2(D_DIM / BN, T_TOK / BM);   // 16 x 16
    for (int e = 0; e < E_EXP; ++e) {
        const float* Wg_e = Wg + (size_t)e * D_DIM * F_DIM;
        const float* Wu_e = Wu + (size_t)e * D_DIM * F_DIM;
        const float* Wd_e = Wd + (size_t)e * F_DIM * D_DIM;
        moe_gateup<<<grid1, 256, 0, stream>>>(x, Wg_e, Wu_e, wTE, hBuf, e);
        moe_down<<<grid2, 256, 0, stream>>>(hBuf, Wd_e, out);
    }
}
